// GAT_12610023981851
// MI455X (gfx1250) — compile-verified, matrix operand order checked by reading, not run on hardware
//
#include <hip/hip_runtime.h>

// -------- problem constants (match reference) --------
#define NNODES 50000
#define NFEAT  256
#define NHID   64
#define NHEADS 8
#define ALPHA  0.2f

typedef __bf16 bf16;
typedef __attribute__((ext_vector_type(16))) __bf16 v16bf;
typedef __attribute__((ext_vector_type(8)))  float  v8f;

union FragBF { v16bf v; unsigned u[8]; };

__device__ __forceinline__ float bf_lo(unsigned u) { return __uint_as_float(u << 16); }
__device__ __forceinline__ float bf_hi(unsigned u) { return __uint_as_float(u & 0xffff0000u); }

__device__ __forceinline__ void atomAddF(float* p, float v) {
    unsafeAtomicAdd(p, v);   // native global_atomic_add_f32 (no CAS loop)
}

// =====================================================================
// Stage A: h_all[head][node][64] (bf16) = x @ W[head], via WMMA bf16
// block = 256 threads (8 waves); each wave -> 16 rows x 64 cols tile
// =====================================================================
__global__ __launch_bounds__(256) void gat_gemm_wmma(
    const float* __restrict__ x, const float* __restrict__ W,
    bf16* __restrict__ h_all)
{
    __shared__ bf16 Xlds[128][34];   // 128 rows x 32 K (padded)
    __shared__ bf16 Wt[64][258];     // W[head] transposed: [n][k] (padded)

    const int head = blockIdx.y;
    const int row0 = blockIdx.x * 128;
    const int t    = threadIdx.x;
    const int lane = t & 31;
    const int wv   = t >> 5;

    // ---- stage W[head] (256x64 f32) into LDS transposed as bf16 ----
    const float* Wh = W + head * (NFEAT * NHID);
    #pragma unroll
    for (int i = 0; i < 64; ++i) {
        int e = i * 256 + t;           // e = k*64 + n  (coalesced read)
        int k = e >> 6, n = e & 63;
        Wt[n][k] = (bf16)Wh[e];
    }
    __syncthreads();

    v8f acc[4] = {};                    // 4 n-blocks of 16x16 f32
    const int mloc  = wv * 16 + (lane & 15);
    const int khalf = (lane & 16) ? 8 : 0;

    for (int kb = 0; kb < NFEAT; kb += 32) {
        // ---- stage 128x32 X tile into LDS as bf16 (coalesced) ----
        {
            int rloc = t >> 1;
            int c0   = (t & 1) * 16;
            int grow = row0 + rloc;
            float4 f[4];
            if (grow < NNODES) {
                const float4* p = (const float4*)(x + (size_t)grow * NFEAT + kb + c0);
                f[0] = p[0]; f[1] = p[1]; f[2] = p[2]; f[3] = p[3];
            } else {
                f[0] = f[1] = f[2] = f[3] = make_float4(0.f, 0.f, 0.f, 0.f);
            }
            #pragma unroll
            for (int q = 0; q < 4; ++q) {
                Xlds[rloc][c0 + 4*q + 0] = (bf16)f[q].x;
                Xlds[rloc][c0 + 4*q + 1] = (bf16)f[q].y;
                Xlds[rloc][c0 + 4*q + 2] = (bf16)f[q].z;
                Xlds[rloc][c0 + 4*q + 3] = (bf16)f[q].w;
            }
        }
        __syncthreads();

        // ---- A fragment: 16x32 bf16, lane L holds row (L&15),
        //      K = {0..7,16..23} (L<16) or {8..15,24..31} (L>=16) ----
        FragBF af;
        #pragma unroll
        for (int i = 0; i < 8; ++i) {
            int k0 = ((i < 4) ? 2*i : 16 + 2*(i - 4)) + khalf;
            af.u[i] = *(const unsigned*)&Xlds[mloc][k0];
        }

        // ---- B fragments (mirror of A with M->N), 4 n-blocks ----
        #pragma unroll
        for (int nb = 0; nb < 4; ++nb) {
            FragBF bfrag;
            int n = nb * 16 + (lane & 15);
            #pragma unroll
            for (int i = 0; i < 8; ++i) {
                int k0 = ((i < 4) ? 2*i : 16 + 2*(i - 4)) + khalf;
                bfrag.u[i] = *(const unsigned*)&Wt[n][kb + k0];
            }
            acc[nb] = __builtin_amdgcn_wmma_f32_16x16x32_bf16(
                false, af.v, false, bfrag.v, (short)0, acc[nb], false, false);
        }
        __syncthreads();
    }

    // ---- store C: VGPR v -> M = v + 8*(lane>=16), N = lane&15 ----
    const int rbase = row0 + wv * 16 + ((lane & 16) ? 8 : 0);
    const int ncol  = lane & 15;
    #pragma unroll
    for (int nb = 0; nb < 4; ++nb) {
        #pragma unroll
        for (int v = 0; v < 8; ++v) {
            int grow = rbase + v;
            if (grow < NNODES)
                h_all[(size_t)(head * NNODES + grow) * NHID + nb * 16 + ncol] =
                    (bf16)acc[nb][v];
        }
    }
}

// =====================================================================
// Stage A2: f1/f2[head][node] = h . a1 / h . a2  (wave per (head,node))
// =====================================================================
__global__ __launch_bounds__(256) void gat_f12(
    const bf16* __restrict__ h_all, const float* __restrict__ a,
    float* __restrict__ f1, float* __restrict__ f2)
{
    int g    = blockIdx.x * 8 + (threadIdx.x >> 5);   // head*N + node
    int lane = threadIdx.x & 31;
    int head = g / NNODES;
    unsigned u = ((const unsigned*)h_all)[(size_t)g * 32 + lane];
    float v0 = bf_lo(u), v1 = bf_hi(u);
    const float* ah = a + head * (2 * NHID);
    float p1 = v0 * ah[2*lane]        + v1 * ah[2*lane + 1];
    float p2 = v0 * ah[NHID + 2*lane] + v1 * ah[NHID + 2*lane + 1];
    #pragma unroll
    for (int off = 16; off; off >>= 1) {
        p1 += __shfl_xor(p1, off);
        p2 += __shfl_xor(p2, off);
    }
    if (lane == 0) { f1[g] = p1; f2[g] = p2; }
}

// =====================================================================
// Stage B: edge scatter, layer 1.  Wave per edge, all 8 heads.
// num[h][src][:] += w * h_all[h][dst][:],  rowsum[h][src] += w
// =====================================================================
__global__ __launch_bounds__(256) void gat_edge1(
    const int* __restrict__ ei, int E,
    const bf16* __restrict__ h_all,
    const float* __restrict__ f1, const float* __restrict__ f2,
    float* __restrict__ rowsum, float* __restrict__ num)
{
    int e = blockIdx.x * 8 + (threadIdx.x >> 5);
    if (e >= E) return;
    int lane = threadIdx.x & 31;
    int src = ei[e], dst = ei[E + e];
    const unsigned* hbits = (const unsigned*)h_all;
    #pragma unroll
    for (int hd = 0; hd < NHEADS; ++hd) {
        float s = f1[hd * NNODES + src] + f2[hd * NNODES + dst];
        float l = (s > 0.f) ? s : ALPHA * s;
        float w = __expf(-l);
        unsigned u = hbits[(size_t)(hd * NNODES + dst) * 32 + lane];
        float* dp = num + (size_t)(hd * NNODES + src) * 64 + 2 * lane;
        atomAddF(dp,     w * bf_lo(u));
        atomAddF(dp + 1, w * bf_hi(u));
        if (lane == 0) atomAddF(rowsum + hd * NNODES + src, w);
    }
}

// =====================================================================
// Stage C: out1 = elu(num/rowsum) fused with 512->1 GEMV (wave per node)
// =====================================================================
__global__ __launch_bounds__(256) void gat_out1_gemv(
    const float* __restrict__ num, const float* __restrict__ rowsum,
    const float* __restrict__ W_out, const float* __restrict__ a_out,
    float* __restrict__ h2, float* __restrict__ f1o, float* __restrict__ f2o)
{
    int node = blockIdx.x * 8 + (threadIdx.x >> 5);
    if (node >= NNODES) return;
    int lane = threadIdx.x & 31;
    float dot = 0.f;
    #pragma unroll
    for (int i = 0; i < 16; ++i) {
        int j = i * 32 + lane;               // 0..511
        int hd = j >> 6, f = j & 63;
        float v = num[(size_t)(hd * NNODES + node) * 64 + f] /
                  rowsum[hd * NNODES + node];
        v = (v > 0.f) ? v : (__expf(v) - 1.f);    // elu
        dot += v * W_out[j];
    }
    #pragma unroll
    for (int off = 16; off; off >>= 1) dot += __shfl_xor(dot, off);
    if (lane == 0) {
        h2[node]  = dot;
        f1o[node] = dot * a_out[0];
        f2o[node] = dot * a_out[1];
    }
}

// =====================================================================
// Stage D: edge scatter, layer 2 (thread per edge; F'=1)
// =====================================================================
__global__ __launch_bounds__(256) void gat_edge2(
    const int* __restrict__ ei, int E,
    const float* __restrict__ h2, const float* __restrict__ f1o,
    const float* __restrict__ f2o,
    float* __restrict__ rowsum2, float* __restrict__ num2)
{
    int e = blockIdx.x * 256 + threadIdx.x;
    if (e >= E) return;
    int src = ei[e], dst = ei[E + e];
    float s = f1o[src] + f2o[dst];
    float l = (s > 0.f) ? s : ALPHA * s;
    float w = __expf(-l);
    atomAddF(rowsum2 + src, w);
    atomAddF(num2 + src, w * h2[dst]);
}

// =====================================================================
// Stage E: out = sigmoid(elu(num2/rowsum2))
// =====================================================================
__global__ __launch_bounds__(256) void gat_final(
    const float* __restrict__ num2, const float* __restrict__ rowsum2,
    float* __restrict__ out)
{
    int n = blockIdx.x * 256 + threadIdx.x;
    if (n >= NNODES) return;
    float v = num2[n] / rowsum2[n];
    v = (v > 0.f) ? v : (__expf(v) - 1.f);
    out[n] = 1.f / (1.f + __expf(-v));
}

// =====================================================================
extern "C" void kernel_launch(void* const* d_in, const int* in_sizes, int n_in,
                              void* d_out, int out_size, void* d_ws, size_t ws_size,
                              hipStream_t stream)
{
    const float* x     = (const float*)d_in[0];
    const int*   ei    = (const int*)  d_in[1];
    const float* W     = (const float*)d_in[2];
    const float* a     = (const float*)d_in[3];
    const float* W_out = (const float*)d_in[4];
    const float* a_out = (const float*)d_in[5];
    const int E = in_sizes[1] / 2;     // 1,650,000

    // ---- carve workspace (aligned 256B) ----
    char*  ws  = (char*)d_ws;
    size_t off = 0;
    auto carve = [&](size_t bytes) -> char* {
        char* p = ws + off;
        off = (off + bytes + 255) & ~(size_t)255;
        return p;
    };
    bf16*  h_all   = (bf16*) carve((size_t)NHEADS * NNODES * NHID * sizeof(bf16));
    float* f1      = (float*)carve((size_t)NHEADS * NNODES * sizeof(float));
    float* f2      = (float*)carve((size_t)NHEADS * NNODES * sizeof(float));
    float* rowsum  = (float*)carve((size_t)NHEADS * NNODES * sizeof(float));
    float* num     = (float*)carve((size_t)NHEADS * NNODES * NHID * sizeof(float));
    float* h2      = (float*)carve((size_t)NNODES * sizeof(float));
    float* f1o     = (float*)carve((size_t)NNODES * sizeof(float));
    float* f2o     = (float*)carve((size_t)NNODES * sizeof(float));
    float* rowsum2 = (float*)carve((size_t)NNODES * sizeof(float));
    float* num2    = (float*)carve((size_t)NNODES * sizeof(float));
    (void)ws_size; (void)n_in; (void)out_size;

    // ---- zero accumulators every call (deterministic replay) ----
    hipMemsetAsync(rowsum,  0, (size_t)NHEADS * NNODES * sizeof(float), stream);
    hipMemsetAsync(num,     0, (size_t)NHEADS * NNODES * NHID * sizeof(float), stream);
    hipMemsetAsync(rowsum2, 0, (size_t)NNODES * sizeof(float), stream);
    hipMemsetAsync(num2,    0, (size_t)NNODES * sizeof(float), stream);

    // ---- pipeline ----
    dim3 gGemm((NNODES + 127) / 128, NHEADS);
    gat_gemm_wmma<<<gGemm, 256, 0, stream>>>(x, W, h_all);

    gat_f12<<<(NHEADS * NNODES) / 8, 256, 0, stream>>>(h_all, a, f1, f2);

    gat_edge1<<<(E + 7) / 8, 256, 0, stream>>>(ei, E, h_all, f1, f2, rowsum, num);

    gat_out1_gemv<<<(NNODES + 7) / 8, 256, 0, stream>>>(num, rowsum, W_out, a_out,
                                                        h2, f1o, f2o);

    gat_edge2<<<(E + 255) / 256, 256, 0, stream>>>(ei, E, h2, f1o, f2o, rowsum2, num2);

    gat_final<<<(NNODES + 255) / 256, 256, 0, stream>>>(num2, rowsum2, (float*)d_out);
}
